// GraphActor_d_46454366273712
// MI455X (gfx1250) — matrix operand order repaired and checked by reading, not compiled, not run on hardware
//
#include <hip/hip_runtime.h>
#include <hip/hip_bf16.h>

// ---------------------------------------------------------------------------
// GraphActor on MI455X (gfx1250, wave32)
//   phase 1: tiny tanh encoders (VALU, memory streaming)
//   phase 2: edge scatter-mean via global_atomic_add_f32 (L2-resident tables)
//   phase 3: actor MLP 48->64->64->1 via v_wmma_f32_16x16x32_f16 (f16 in, f32 acc)
// ---------------------------------------------------------------------------

typedef __attribute__((ext_vector_type(16))) _Float16 v16h;
typedef __attribute__((ext_vector_type(8)))  _Float16 v8h;
typedef __attribute__((ext_vector_type(8)))  float    v8f;

// ---------------- encoders ----------------

template<int D>
__global__ void encode_f32_k(const float* __restrict__ X,
                             const float* __restrict__ W,
                             const float* __restrict__ B,
                             float* __restrict__ out, int N)
{
    int r = blockIdx.x * blockDim.x + threadIdx.x;
    if (r >= N) return;
    float x[D];
#pragma unroll
    for (int k = 0; k < D; ++k) x[k] = X[(size_t)r * D + k];
#pragma unroll
    for (int j = 0; j < 16; ++j) {
        float a = B[j];
#pragma unroll
        for (int k = 0; k < D; ++k) a += x[k] * W[k * 16 + j];
        out[(size_t)r * 16 + j] = tanhf(a);
    }
}

// request encoder writes f16 directly into action-feature cols [0,16)
template<int D>
__global__ void encode_req_k(const float* __restrict__ X,
                             const float* __restrict__ W,
                             const float* __restrict__ B,
                             _Float16* __restrict__ AF, int N)
{
    int r = blockIdx.x * blockDim.x + threadIdx.x;
    if (r >= N) return;
    float x[D];
#pragma unroll
    for (int k = 0; k < D; ++k) x[k] = X[(size_t)r * D + k];
#pragma unroll
    for (int j = 0; j < 16; ++j) {
        float a = B[j];
#pragma unroll
        for (int k = 0; k < D; ++k) a += x[k] * W[k * 16 + j];
        AF[(size_t)r * 64 + j] = (_Float16)tanhf(a);
    }
}

__global__ void zero_pad_rows_k(_Float16* __restrict__ AF, int n0, int n1)
{
    int idx = blockIdx.x * blockDim.x + threadIdx.x;
    int r = n0 + (idx >> 6), j = idx & 63;
    if (r < n1) AF[(size_t)r * 64 + j] = (_Float16)0.0f;
}

// ---------------- edge scatter-mean ----------------

// pas_feat[recv[e]] accumulated onto vehicle send[e]  (16 floats/edge)
__global__ void scatter_vp_k(const int* __restrict__ recv, const int* __restrict__ send,
                             const float* __restrict__ pasF,
                             float* __restrict__ pool, float* __restrict__ cnt, int E)
{
    int e = blockIdx.x * blockDim.x + threadIdx.x;
    if (e >= E) return;
    int p = recv[e], v = send[e];
    const float4* src = (const float4*)(pasF + (size_t)p * 16);
    float* dst = pool + (size_t)v * 16;
#pragma unroll
    for (int q = 0; q < 4; ++q) {
        float4 f = src[q];
        atomicAdd(dst + q * 4 + 0, f.x);
        atomicAdd(dst + q * 4 + 1, f.y);
        atomicAdd(dst + q * 4 + 2, f.z);
        atomicAdd(dst + q * 4 + 3, f.w);
    }
    atomicAdd(cnt + v, 1.0f);
}

__global__ void finalize_veh_k(const float* __restrict__ vehF,
                               const float* __restrict__ pool,
                               const float* __restrict__ cnt,
                               float* __restrict__ veh32, int total)
{
    int idx = blockIdx.x * blockDim.x + threadIdx.x;
    if (idx >= total) return;
    int v = idx >> 5, j = idx & 31;
    float val = (j < 16) ? vehF[(size_t)v * 16 + j]
                         : pool[(size_t)v * 16 + (j - 16)] / fmaxf(cnt[v], 1.0f);
    veh32[idx] = val;
}

// veh32[recv[e]] accumulated onto request send[e]  (32 floats/edge)
__global__ void scatter_rv_k(const int* __restrict__ recv, const int* __restrict__ send,
                             const float* __restrict__ veh32,
                             float* __restrict__ acc, float* __restrict__ cnt, int E)
{
    int e = blockIdx.x * blockDim.x + threadIdx.x;
    if (e >= E) return;
    int v = recv[e], r = send[e];
    const float4* src = (const float4*)(veh32 + (size_t)v * 32);
    float* dst = acc + (size_t)r * 32;
#pragma unroll
    for (int q = 0; q < 8; ++q) {
        float4 f = src[q];
        atomicAdd(dst + q * 4 + 0, f.x);
        atomicAdd(dst + q * 4 + 1, f.y);
        atomicAdd(dst + q * 4 + 2, f.z);
        atomicAdd(dst + q * 4 + 3, f.w);
    }
    atomicAdd(cnt + r, 1.0f);
}

// write action-feature cols [16,64): 32 aggregated cols + 16 zero-pad cols (f16)
__global__ void finalize_req_k(const float* __restrict__ acc,
                               const float* __restrict__ cnt,
                               _Float16* __restrict__ AF, int N)
{
    long long idx = (long long)blockIdx.x * blockDim.x + threadIdx.x;
    if (idx >= (long long)N * 48) return;
    int r = (int)(idx / 48), j = (int)(idx % 48);
    float val = 0.0f;
    if (j < 32) val = acc[(size_t)r * 32 + j] / fmaxf(cnt[r], 1.0f);
    AF[(size_t)r * 64 + 16 + j] = (_Float16)val;
}

// ---------------- actor MLP via WMMA ----------------
// A-frag layout (16-bit A 16x32): lane L: row = L&15, h = L>>4,
//   elem i -> K = h*8 + (i&7) + (i>=8 ? 16 : 0)   => two contiguous v8h loads.
// B-frag layout (16-bit B 32x16): lane L: col = L&15, h = L>>4,
//   elem i -> K = h*8 + (i&7) + (i>=8 ? 16 : 0), same as A                         => pre-swizzled, one v16h load.
// C/D f32 16x16: lane L, vgpr g -> row = g + 8*(L>>4), col = L&15.

__launch_bounds__(256)
__global__ void actor_mlp_k(const _Float16* __restrict__ AF,
                            const float* __restrict__ gW1, const float* __restrict__ gB1,
                            const float* __restrict__ gW2, const float* __restrict__ gB2,
                            const float* __restrict__ gW3, const float* __restrict__ gB3,
                            float* __restrict__ out, int nrows, int ntiles)
{
    __shared__ _Float16 sW1[2][4][32][16];   // [kchunk][coltile][lane][16]  8KB
    __shared__ _Float16 sW2[2][4][32][16];   //                              8KB
    __shared__ float    sB1[64], sB2[64], sW3[64];
    __shared__ _Float16 sH[8][16][64];       // per-wave layer1 staging     16KB

    const int tid = threadIdx.x;

    // cooperative weight swizzle into B-fragment layout (K padded 48->64 for W1)
    for (int idx = tid; idx < 4096; idx += 256) {
        int i  = idx & 15;
        int ln = (idx >> 4) & 31;
        int t  = (idx >> 9) & 3;
        int kk = (idx >> 11) & 1;
        int n = ln & 15, h = ln >> 4;
        int K = kk * 32 + h * 8 + (i & 7) + ((i >= 8) ? 16 : 0);
        int col = t * 16 + n;
        sW1[kk][t][ln][i] = (_Float16)((K < 48) ? gW1[K * 64 + col] : 0.0f);
        sW2[kk][t][ln][i] = (_Float16)gW2[K * 64 + col];
    }
    if (tid < 64) { sB1[tid] = gB1[tid]; sB2[tid] = gB2[tid]; sW3[tid] = gW3[tid]; }
    __syncthreads();

    const int wv = tid >> 5;
    const int lane = tid & 31;
    const int n = lane & 15, h = lane >> 4;
    const int tile = blockIdx.x * 8 + wv;
    if (tile >= ntiles) return;
    const int row0 = tile * 16;

    union V16 { v16h v; v8h p[2]; };

    // ---- layer 1: A from global f16 action features (K = 64 incl. zero pad)
    const _Float16* Arow = AF + (size_t)(row0 + n) * 64;
    V16 a0, a1;
    a0.p[0] = *(const v8h*)(Arow + h * 8);
    a0.p[1] = *(const v8h*)(Arow + 16 + h * 8);
    a1.p[0] = *(const v8h*)(Arow + 32 + h * 8);
    a1.p[1] = *(const v8h*)(Arow + 48 + h * 8);

    v8f acc[4];
#pragma unroll
    for (int t = 0; t < 4; ++t) {
        float bias = sB1[t * 16 + n];
        v8f c;
#pragma unroll
        for (int g = 0; g < 8; ++g) c[g] = bias;
        v16h b0 = *(const v16h*)(&sW1[0][t][lane][0]);
        v16h b1 = *(const v16h*)(&sW1[1][t][lane][0]);
        c = __builtin_amdgcn_wmma_f32_16x16x32_f16(false, a0.v, false, b0, (short)0, c, false, false);
        c = __builtin_amdgcn_wmma_f32_16x16x32_f16(false, a1.v, false, b1, (short)0, c, false, false);
        acc[t] = c;
    }

    // tanh + restage as f16 A-matrix for layer 2 (intra-wave LDS, dscnt-ordered)
#pragma unroll
    for (int t = 0; t < 4; ++t)
#pragma unroll
        for (int g = 0; g < 8; ++g)
            sH[wv][g + 8 * h][t * 16 + n] = (_Float16)tanhf(acc[t][g]);

    // ---- layer 2: A from LDS staging (K = 64)
    const _Float16* Hrow = &sH[wv][n][0];
    a0.p[0] = *(const v8h*)(Hrow + h * 8);
    a0.p[1] = *(const v8h*)(Hrow + 16 + h * 8);
    a1.p[0] = *(const v8h*)(Hrow + 32 + h * 8);
    a1.p[1] = *(const v8h*)(Hrow + 48 + h * 8);

#pragma unroll
    for (int t = 0; t < 4; ++t) {
        float bias = sB2[t * 16 + n];
        v8f c;
#pragma unroll
        for (int g = 0; g < 8; ++g) c[g] = bias;
        v16h b0 = *(const v16h*)(&sW2[0][t][lane][0]);
        v16h b1 = *(const v16h*)(&sW2[1][t][lane][0]);
        c = __builtin_amdgcn_wmma_f32_16x16x32_f16(false, a0.v, false, b0, (short)0, c, false, false);
        c = __builtin_amdgcn_wmma_f32_16x16x32_f16(false, a1.v, false, b1, (short)0, c, false, false);
        acc[t] = c;
    }

    // ---- layer 3: out[r] = sum_col tanh(h2[r][col]) * W3[col] + b3
    float p[8];
#pragma unroll
    for (int g = 0; g < 8; ++g) p[g] = 0.0f;
#pragma unroll
    for (int t = 0; t < 4; ++t) {
        float w3 = sW3[t * 16 + n];
#pragma unroll
        for (int g = 0; g < 8; ++g) p[g] += tanhf(acc[t][g]) * w3;
    }
    // butterfly reduce across the 16 lanes of each half-wave (wave32)
#pragma unroll
    for (int m = 1; m < 16; m <<= 1)
#pragma unroll
        for (int g = 0; g < 8; ++g) p[g] += __shfl_xor(p[g], m, 32);

    if (n == 0) {
        float b3 = gB3[0];
#pragma unroll
        for (int g = 0; g < 8; ++g) {
            int r = row0 + g + 8 * h;
            if (r < nrows) out[r] = p[g] + b3;
        }
    }
}

// ---------------- host ----------------

extern "C" void kernel_launch(void* const* d_in, const int* in_sizes, int n_in,
                              void* d_out, int out_size, void* d_ws, size_t ws_size,
                              hipStream_t stream)
{
    const int D_REQ = 10, D_VEH = 8, D_PAS = 10;
    const int N_REQ = in_sizes[0] / D_REQ;
    const int N_VEH = in_sizes[1] / D_VEH;
    const int N_PAS = in_sizes[2] / D_PAS;
    const int E_VP  = in_sizes[3];
    const int E_RV  = in_sizes[5];
    const int NR_PAD = ((N_REQ + 15) / 16) * 16;

    const float* reqX = (const float*)d_in[0];
    const float* vehX = (const float*)d_in[1];
    const float* pasX = (const float*)d_in[2];
    const int* vp_recv = (const int*)d_in[3];
    const int* vp_send = (const int*)d_in[4];
    const int* rv_send = (const int*)d_in[5];
    const int* rv_recv = (const int*)d_in[6];
    const float* W_req = (const float*)d_in[7];  const float* b_req = (const float*)d_in[8];
    const float* W_veh = (const float*)d_in[9];  const float* b_veh = (const float*)d_in[10];
    const float* W_pas = (const float*)d_in[11]; const float* b_pas = (const float*)d_in[12];
    const float* W1 = (const float*)d_in[13]; const float* b1 = (const float*)d_in[14];
    const float* W2 = (const float*)d_in[15]; const float* b2 = (const float*)d_in[16];
    const float* W3 = (const float*)d_in[17]; const float* b3 = (const float*)d_in[18];
    float* out = (float*)d_out;

    // workspace carve-out (256B aligned)
    char* ws = (char*)d_ws;
    size_t off = 0;
    auto carve = [&](size_t bytes) -> char* {
        char* p = ws + off;
        off = (off + bytes + 255) & ~(size_t)255;
        return p;
    };
    _Float16* AF     = (_Float16*)carve((size_t)NR_PAD * 64 * sizeof(_Float16));
    float* pasF      = (float*)carve((size_t)N_PAS * 16 * sizeof(float));
    float* vehF      = (float*)carve((size_t)N_VEH * 16 * sizeof(float));
    float* veh32     = (float*)carve((size_t)N_VEH * 32 * sizeof(float));
    char*  zbase     = ws + off;
    float* vehPool   = (float*)carve((size_t)N_VEH * 16 * sizeof(float));
    float* vehCnt    = (float*)carve((size_t)N_VEH * sizeof(float));
    float* reqAcc    = (float*)carve((size_t)N_REQ * 32 * sizeof(float));
    float* reqCnt    = (float*)carve((size_t)N_REQ * sizeof(float));
    size_t zbytes = (size_t)((ws + off) - zbase);

    hipMemsetAsync(zbase, 0, zbytes, stream);

    const int B = 256;
    encode_req_k<10><<<(N_REQ + B - 1) / B, B, 0, stream>>>(reqX, W_req, b_req, AF, N_REQ);
    encode_f32_k<8><<<(N_VEH + B - 1) / B, B, 0, stream>>>(vehX, W_veh, b_veh, vehF, N_VEH);
    encode_f32_k<10><<<(N_PAS + B - 1) / B, B, 0, stream>>>(pasX, W_pas, b_pas, pasF, N_PAS);
    if (NR_PAD > N_REQ) {
        int padElems = (NR_PAD - N_REQ) * 64;
        zero_pad_rows_k<<<(padElems + B - 1) / B, B, 0, stream>>>(AF, N_REQ, NR_PAD);
    }

    scatter_vp_k<<<(E_VP + B - 1) / B, B, 0, stream>>>(vp_recv, vp_send, pasF, vehPool, vehCnt, E_VP);
    finalize_veh_k<<<(N_VEH * 32 + B - 1) / B, B, 0, stream>>>(vehF, vehPool, vehCnt, veh32, N_VEH * 32);
    scatter_rv_k<<<(E_RV + B - 1) / B, B, 0, stream>>>(rv_recv, rv_send, veh32, reqAcc, reqCnt, E_RV);
    long long fin = (long long)N_REQ * 48;
    finalize_req_k<<<(int)((fin + B - 1) / B), B, 0, stream>>>(reqAcc, reqCnt, AF, N_REQ);

    int ntiles = NR_PAD / 16;
    actor_mlp_k<<<(ntiles + 7) / 8, B, 0, stream>>>(AF, W1, b1, W2, b2, W3, b3, out, N_REQ, ntiles);
}
